// TopKRouter_61675730370567
// MI455X (gfx1250) — matrix operand order repaired and checked by reading, not compiled, not run on hardware
//
#include <hip/hip_runtime.h>

typedef __attribute__((ext_vector_type(16))) __bf16 v16bf;
typedef __attribute__((ext_vector_type(8)))  float  v8f;
typedef __attribute__((ext_vector_type(4)))  float  f4;

#define TOKENS   16384
#define DIM      2048
#define NEXP     64
#define M_PER_WG 64   // tokens per workgroup (4 token-tiles x 16)

__global__ __launch_bounds__(256)
void topk_router_kernel(const float* __restrict__ x,
                        const float* __restrict__ W,
                        const float* __restrict__ bias,
                        float* __restrict__ out_probs,
                        int*   __restrict__ out_idx,
                        float* __restrict__ out_logits)
{
    __shared__ float lds_logits[M_PER_WG * NEXP];   // 16 KB

    const int tid    = threadIdx.x;
    const int lane   = tid & 31;
    const int wave   = tid >> 5;
    const int n_half = wave & 1;    // which 32-expert half
    const int m_tile = wave >> 1;   // which 16-token tile within the block (0..3)
    const int wg_tok = blockIdx.x * M_PER_WG;

    const int  lane_lo = lane & 15;
    const bool hi      = lane >= 16;

    const int a_row = wg_tok + m_tile * 16 + lane_lo;   // token this lane feeds as A-row
    const int e0    = n_half * 32 + lane_lo;            // expert (B-col) for tile 0
    const int e1    = e0 + 16;                          // expert (B-col) for tile 1

    // A k order used here (contiguous, matches B): lane<16 -> K 0..15; lane>=16 -> K 16..31
    const float* xrow  = x + (size_t)a_row * DIM + (hi ? 16 : 0);
    // B k order used here (contiguous, matches A): lane<16 -> K 0..15 of column e; lane>=16 -> K 16..31
    const float* wrow0 = W + (size_t)e0 * DIM + (hi ? 16 : 0);
    const float* wrow1 = W + (size_t)e1 * DIM + (hi ? 16 : 0);

    v8f c0 = {};
    v8f c1 = {};

    #pragma unroll 2
    for (int k0 = 0; k0 < DIM; k0 += 32) {
        // A fragment: loaded ONCE, feeds both WMMAs
        f4 a0 = *(const f4*)(xrow + k0);
        f4 a1 = *(const f4*)(xrow + k0 + 4);
        f4 a2 = *(const f4*)(xrow + k0 + 8);
        f4 a3 = *(const f4*)(xrow + k0 + 12);

        // B fragments: 16 contiguous W values per lane per tile
        f4 p0 = *(const f4*)(wrow0 + k0);
        f4 p1 = *(const f4*)(wrow0 + k0 + 4);
        f4 p2 = *(const f4*)(wrow0 + k0 + 8);
        f4 p3 = *(const f4*)(wrow0 + k0 + 12);

        f4 q0 = *(const f4*)(wrow1 + k0);
        f4 q1 = *(const f4*)(wrow1 + k0 + 4);
        f4 q2 = *(const f4*)(wrow1 + k0 + 8);
        f4 q3 = *(const f4*)(wrow1 + k0 + 12);

        v16bf a, b0, b1;
        #pragma unroll
        for (int i = 0; i < 4; ++i) {
            a[i]       = (__bf16)a0[i];
            a[i + 4]   = (__bf16)a1[i];
            a[i + 8]   = (__bf16)a2[i];
            a[i + 12]  = (__bf16)a3[i];
            b0[i]      = (__bf16)p0[i];
            b0[i + 4]  = (__bf16)p1[i];
            b0[i + 8]  = (__bf16)p2[i];
            b0[i + 12] = (__bf16)p3[i];
            b1[i]      = (__bf16)q0[i];
            b1[i + 4]  = (__bf16)q1[i];
            b1[i + 8]  = (__bf16)q2[i];
            b1[i + 12] = (__bf16)q3[i];
        }
        // D = A(16x32 bf16) x B(32x16 bf16) + C(f32), two expert tiles per A
        c0 = __builtin_amdgcn_wmma_f32_16x16x32_bf16(false, a, false, b0,
                                                     (short)0, c0, false, false);
        c1 = __builtin_amdgcn_wmma_f32_16x16x32_bf16(false, a, false, b1,
                                                     (short)0, c1, false, false);
    }

    // C layout: VGPR j, lanes 0-15 -> (M=j, N=lane); lanes 16-31 -> (M=j+8, N=lane-16)
    const float bv0 = bias[e0];
    const float bv1 = bias[e1];
    #pragma unroll
    for (int j = 0; j < 8; ++j) {
        const int tok_local = m_tile * 16 + j + (hi ? 8 : 0);
        const float v0 = c0[j] + bv0;
        const float v1 = c1[j] + bv1;
        float* grow = out_logits + (size_t)(wg_tok + tok_local) * NEXP;
        grow[e0] = v0;
        grow[e1] = v1;
        lds_logits[tok_local * NEXP + e0] = v0;
        lds_logits[tok_local * NEXP + e1] = v1;
    }
    __syncthreads();

    // Top-2 + softmax over the two kept logits; one lane per token.
    if (tid < M_PER_WG) {
        const float* row = lds_logits + tid * NEXP;
        float v1 = row[0]; int i1 = 0;
        float v2 = -__builtin_inff(); int i2 = -1;
        for (int n = 1; n < NEXP; ++n) {
            const float v = row[n];
            if (v > v1)      { v2 = v1; i2 = i1; v1 = v; i1 = n; }
            else if (v > v2) { v2 = v;  i2 = n; }
        }
        const float e2  = __expf(v2 - v1);
        const float inv = 1.0f / (1.0f + e2);
        const int t = wg_tok + tid;
        out_probs[t * 2]     = inv;
        out_probs[t * 2 + 1] = e2 * inv;
        out_idx[t * 2]       = i1;
        out_idx[t * 2 + 1]   = i2;
    }
}

extern "C" void kernel_launch(void* const* d_in, const int* in_sizes, int n_in,
                              void* d_out, int out_size, void* d_ws, size_t ws_size,
                              hipStream_t stream) {
    const float* x = (const float*)d_in[0];   // [16384, 2048]
    const float* W = (const float*)d_in[1];   // [64, 2048]
    const float* b = (const float*)d_in[2];   // [64]

    float* probs  = (float*)d_out;                       // [16384, 2] f32
    int*   idx    = (int*)(probs + (size_t)TOKENS * 2);  // [16384, 2] i32 (bit region)
    float* logits = (float*)d_out + (size_t)TOKENS * 4;  // [16384, 64] f32

    topk_router_kernel<<<TOKENS / M_PER_WG, 256, 0, stream>>>(x, W, b, probs, idx, logits);
}
